// Ladder_52158082843220
// MI455X (gfx1250) — matrix operand order repaired and checked by reading, not compiled, not run on hardware
//
#include <hip/hip_runtime.h>
#include <hip/hip_bf16.h>

// ---------------------------------------------------------------------------
// Ladder transformer forward for MI455X (gfx1250, wave32, WMMA).
//   * one bf16-WMMA GEMM kernel: 128x128x32 block tile, 256 threads / 8 waves,
//     each wave computes 32x64 via 2x4 v_wmma_f32_16x16x32_bf16
//   * DOUBLE-BUFFERED LDS pipeline: async bf16 B copies + fp32->bf16 A staging
//     for tile t+1 overlap the WMMA compute of tile t; one barrier per K-step
//   * B is always bf16 [N,K] K-contiguous, staged with the gfx1250 async path:
//     global_load_async_to_lds_b128 + s_wait_asynccnt
//   * weights converted fp32->bf16 once into scratch slots; K/V projection
//     epilogues emit bf16 directly (V transposed) so attention GEMMs also use
//     the async bf16 B path
//   * fused epilogue: bias / scale / causal+pad mask / relu / residual,
//     output f32, bf16, or bf16-transposed
// ---------------------------------------------------------------------------

typedef __attribute__((ext_vector_type(16))) __bf16 v16bf;
typedef __attribute__((ext_vector_type(8)))  __bf16 v8bf;
typedef __attribute__((ext_vector_type(8)))  float  v8f;

#define BLK_M 128
#define BLK_N 128
#define BLK_K 32
#define LDSS  40   // BLK_K + 8 halves padding: 80B row stride, 16B aligned

#define CAT16(lo, hi) \
  __builtin_shufflevector(lo, hi, 0,1,2,3,4,5,6,7,8,9,10,11,12,13,14,15)
#define WMMA_BF16(Af, Bf, Cf) \
  __builtin_amdgcn_wmma_f32_16x16x32_bf16(false, (Af), false, (Bf), (short)0, (Cf), false, false)

#define OUT_F32    0
#define OUT_BF16   1
#define OUT_BF16_T 2   // C16[b][n][s] transposed per-batch (for V)

__global__ __launch_bounds__(256)
void gemm_wmma(const float*  __restrict__ A,      // [M,K] fp32
               const __bf16* __restrict__ Bw,     // [N,K] bf16, K contiguous
               const float*  __restrict__ bias,   // [N] or null
               const float*  __restrict__ res,    // residual [M,N] or null
               float*        __restrict__ Cf,     // f32 out (OUT_F32)
               __bf16*       __restrict__ C16,    // bf16 out (OUT_BF16*)
               int M, int N, int K,
               long long aBatch, long long bBatch,
               long long cBatch, long long rBatch,
               float scale, int relu,
               const int* __restrict__ mask, long long maskBatch,
               int causal, int outmode, int Sdim)
{
    // double-buffered tiles (ping-pong)
    __shared__ __attribute__((aligned(16))) __bf16 As[2][BLK_M][LDSS];
    __shared__ __attribute__((aligned(16))) __bf16 Bs[2][BLK_N][LDSS];

    const int tid   = threadIdx.x;
    const int lane  = tid & 31;
    const int wave  = tid >> 5;          // 0..7
    const int waveM = (wave >> 1) * 32;  // 0,32,64,96
    const int waveN = (wave & 1) * 64;   // 0,64
    const int half  = lane >> 4;
    const int l16   = lane & 15;

    const int bz = blockIdx.z;
    const int m0 = blockIdx.y * BLK_M;
    const int n0 = blockIdx.x * BLK_N;

    const float*  Ab = A  + (size_t)aBatch * bz;
    const __bf16* Bb = Bw + (size_t)bBatch * bz;

    // ---- staging work split ----
    const int ar = tid >> 1;         // 0..127  A row
    const int ac = (tid & 1) * 16;   // A col base (16 floats = 4x float4)
    const int br0 = tid >> 2;        // 0..63   B row, chunk 0
    const int br1 = br0 + 64;        //         B row, chunk 1
    const int bo  = (tid & 3) * 8;   // B col base (8 bf16 = 16B)

    const unsigned ldsB0[2] = { (unsigned)(uintptr_t)&Bs[0][br0][bo],
                                (unsigned)(uintptr_t)&Bs[1][br0][bo] };
    const unsigned ldsB1[2] = { (unsigned)(uintptr_t)&Bs[0][br1][bo],
                                (unsigned)(uintptr_t)&Bs[1][br1][bo] };

    // issue async bf16 B tile copy (gfx1250 ASYNCcnt path)
    auto issueB = [&](int kb, int buf) {
        const __bf16* g0 = Bb + (size_t)(n0 + br0) * K + kb + bo;
        const __bf16* g1 = Bb + (size_t)(n0 + br1) * K + kb + bo;
        asm volatile("global_load_async_to_lds_b128 %0, %1, off"
                     :: "v"(ldsB0[buf]), "v"((unsigned long long)(uintptr_t)g0)
                     : "memory");
        asm volatile("global_load_async_to_lds_b128 %0, %1, off"
                     :: "v"(ldsB1[buf]), "v"((unsigned long long)(uintptr_t)g1)
                     : "memory");
    };
    // stage A tile fp32 -> bf16 through VGPRs
    auto stageA = [&](int kb, int buf) {
        const float4* src = (const float4*)(Ab + (size_t)(m0 + ar) * K + kb + ac);
        #pragma unroll
        for (int j = 0; j < 4; ++j) {
            float4 f = src[j];
            __bf16* d = &As[buf][ar][ac + 4 * j];
            d[0] = (__bf16)f.x; d[1] = (__bf16)f.y;
            d[2] = (__bf16)f.z; d[3] = (__bf16)f.w;
        }
    };

    v8f acc[2][4] = {};
    const int nT = K / BLK_K;

    // ---- prologue: fill buffer 0 ----
    issueB(0, 0);
    stageA(0, 0);

    for (int t = 0; t < nT; ++t) {
        const int cur = t & 1;
        const int nxt = cur ^ 1;
        const int kb_next = (t + 1) * BLK_K;

        // tile t's async copies must have landed; barrier also publishes the
        // A-stores of tile t and retires all reads of the buffer we refill.
        asm volatile("s_wait_asynccnt 0x0" ::: "memory");
        __syncthreads();

        if (t + 1 < nT) issueB(kb_next, nxt);

        // ---- fragments (ISA VGPR layouts) ----
        // A 16x32: lanes 0-15 K{0..7,16..23}; lanes 16-31 K{8..15,24..31}
        v16bf afr[2];
        #pragma unroll
        for (int i = 0; i < 2; ++i) {
            const int r = waveM + i * 16 + l16;
            v8bf lo = *(const v8bf*)&As[cur][r][half * 8];
            v8bf hi = *(const v8bf*)&As[cur][r][16 + half * 8];
            afr[i] = CAT16(lo, hi);
        }
        // B 32x16: lane = column N=l16; lanes 0-15 K{0..7,16..23}; lanes 16-31 K{8..15,24..31}
        v16bf bfr[4];
        #pragma unroll
        for (int j = 0; j < 4; ++j) {
            const int c = waveN + j * 16 + l16;
            v8bf lo = *(const v8bf*)&Bs[cur][c][half * 8];
            v8bf hi = *(const v8bf*)&Bs[cur][c][16 + half * 8];
            bfr[j] = CAT16(lo, hi);
        }

        #pragma unroll
        for (int i = 0; i < 2; ++i)
            #pragma unroll
            for (int j = 0; j < 4; ++j)
                acc[i][j] = WMMA_BF16(afr[i], bfr[j], acc[i][j]);

        // overlap next A staging (global load + cvt + ds_store) with WMMAs
        if (t + 1 < nT) stageA(kb_next, nxt);
    }

    // ---- epilogue ----
    const int* Mb = mask ? mask + (size_t)maskBatch * bz : nullptr;

    #pragma unroll
    for (int i = 0; i < 2; ++i) {
        #pragma unroll
        for (int j = 0; j < 4; ++j) {
            const int gcol  = n0 + waveN + j * 16 + l16;
            const int rbase = m0 + waveM + i * 16 + half * 8;
            #pragma unroll
            for (int r = 0; r < 8; ++r) {
                const int grow = rbase + r;
                float v = acc[i][j][r];
                if (bias) v += bias[gcol];
                v *= scale;
                if (Mb) {
                    const bool keep = (Mb[gcol] != 0) && (!causal || gcol <= grow);
                    if (!keep) v = -1.0e9f;
                }
                if (relu) v = fmaxf(v, 0.0f);
                if (outmode == OUT_F32) {
                    if (res) v += res[(size_t)rBatch * bz + (size_t)grow * N + gcol];
                    Cf[(size_t)cBatch * bz + (size_t)grow * N + gcol] = v;
                } else if (outmode == OUT_BF16) {
                    C16[(size_t)cBatch * bz + (size_t)grow * N + gcol] = (__bf16)v;
                } else { // OUT_BF16_T : [b][n][s]
                    const int b = grow / Sdim;
                    const int s = grow - b * Sdim;
                    C16[(size_t)b * N * Sdim + (size_t)gcol * Sdim + s] = (__bf16)v;
                }
            }
        }
    }
}

// Row-wise softmax over S elements; one block per row, wave32 reductions.
__global__ __launch_bounds__(256)
void softmax_rows(float* __restrict__ p, int S)
{
    float* x = p + (size_t)blockIdx.x * S;
    __shared__ float red[8];
    const int tid = threadIdx.x;
    const int w   = tid >> 5;

    float m = -3.4e38f;
    for (int i = tid; i < S; i += 256) m = fmaxf(m, x[i]);
    #pragma unroll
    for (int off = 16; off; off >>= 1) m = fmaxf(m, __shfl_xor(m, off, 32));
    if ((tid & 31) == 0) red[w] = m;
    __syncthreads();
    float mx = red[0];
    #pragma unroll
    for (int j = 1; j < 8; ++j) mx = fmaxf(mx, red[j]);
    __syncthreads();

    float s = 0.0f;
    for (int i = tid; i < S; i += 256) {
        float e = __expf(x[i] - mx);
        x[i] = e;
        s += e;
    }
    #pragma unroll
    for (int off = 16; off; off >>= 1) s += __shfl_xor(s, off, 32);
    if ((tid & 31) == 0) red[w] = s;
    __syncthreads();
    float tot = 0.0f;
    #pragma unroll
    for (int j = 0; j < 8; ++j) tot += red[j];
    const float inv = 1.0f / tot;
    for (int i = tid; i < S; i += 256) x[i] *= inv;
}

// Bulk fp32 -> bf16 conversion (weights), 8 elems / thread, 16B stores.
__global__ __launch_bounds__(256)
void f32_to_bf16(const float* __restrict__ in, __bf16* __restrict__ out,
                 long long n)
{
    long long i = ((long long)blockIdx.x * 256 + threadIdx.x) * 8;
    if (i + 8 <= n) {
        float4 f0 = *(const float4*)(in + i);
        float4 f1 = *(const float4*)(in + i + 4);
        v8bf o;
        o[0] = (__bf16)f0.x; o[1] = (__bf16)f0.y;
        o[2] = (__bf16)f0.z; o[3] = (__bf16)f0.w;
        o[4] = (__bf16)f1.x; o[5] = (__bf16)f1.y;
        o[6] = (__bf16)f1.z; o[7] = (__bf16)f1.w;
        *(v8bf*)(out + i) = o;
    }
}

// ---------------------------------------------------------------------------

enum { cH0 = 2048, cH1 = 1024, cNL = 8, cB = 2, cS = 2048, cH4 = 4096 };

static inline void conv_w(const float* w, __bf16* out, long long n, hipStream_t s)
{
    f32_to_bf16<<<dim3((unsigned)(n / 8 / 256)), dim3(256), 0, s>>>(w, out, n);
}

static inline void launch_gemm(const float* A, const __bf16* Bw,
                               const float* bias, const float* res,
                               float* Cf, __bf16* C16,
                               int M, int N, int K,
                               long long aB, long long bB,
                               long long cB_, long long rB,
                               int batches, float scale, int relu,
                               const int* mask, long long maskB, int causal,
                               int outmode, int Sdim, hipStream_t s)
{
    dim3 grid(N / BLK_N, M / BLK_M, batches);
    dim3 block(256);
    gemm_wmma<<<grid, block, 0, s>>>(A, Bw, bias, res, Cf, C16, M, N, K,
                                     aB, bB, cB_, rB, scale, relu,
                                     mask, maskB, causal, outmode, Sdim);
}

extern "C" void kernel_launch(void* const* d_in, const int* in_sizes, int n_in,
                              void* d_out, int out_size, void* d_ws, size_t ws_size,
                              hipStream_t stream)
{
    (void)in_sizes; (void)n_in; (void)out_size; (void)ws_size;

    const float* z       = (const float*)d_in[0];
    const float* act0    = (const float*)d_in[1];
    const float* act1    = (const float*)d_in[2];
    const int*   attmask = (const int*)  d_in[3];
    const float* dproj_w = (const float*)d_in[4];
    const float* dproj_b = (const float*)d_in[5];
    const float* uproj_w = (const float*)d_in[6];
    const float* uproj_b = (const float*)d_in[7];
    const float* wq_w    = (const float*)d_in[8];
    const float* wq_b    = (const float*)d_in[9];
    const float* wk_w    = (const float*)d_in[10];
    const float* wk_b    = (const float*)d_in[11];
    const float* wv_w    = (const float*)d_in[12];
    const float* wv_b    = (const float*)d_in[13];
    const float* mlp1_w  = (const float*)d_in[14];
    const float* mlp1_b  = (const float*)d_in[15];
    const float* mlp2_w  = (const float*)d_in[16];
    const float* mlp2_b  = (const float*)d_in[17];

    float* out = (float*)d_out;

    const size_t tok = (size_t)cB * cS;              // 4096 tokens
    // f32 region
    float* ws  = (float*)d_ws;
    float* rs  = ws;                                 // tok*H1      (4M)
    float* q   = rs  + tok * cH1;                    // tok*H1      (4M)
    float* sc  = q   + tok * cH1;                    // B*S*S       (8M)
    float* hid = sc  + (size_t)cB * cS * cS;         // tok*4H1     (16M)
    // bf16 region
    __bf16* k16 = (__bf16*)(hid + tok * cH4);        // tok*H1
    __bf16* v16 = k16 + tok * cH1;                   // B*H1*S (transposed V)
    __bf16* w0  = v16 + tok * cH1;                   // weight slot 0 (4M)
    __bf16* w1  = w0  + (size_t)cH4 * cH1;           // weight slot 1 (4M)

    const int   M  = (int)tok;
    const float att_scale = 0.03125f;                // 1/sqrt(H1)

    // rs = dproj(act0)
    conv_w(dproj_w, w0, (long long)cH1 * cH0, stream);
    launch_gemm(act0, w0, dproj_b, nullptr, rs, nullptr, M, cH1, cH0,
                0, 0, 0, 0, 1, 1.0f, 0, nullptr, 0, 0, OUT_F32, cS, stream);

    for (int i = 0; i < cNL; ++i) {
        if (i == 1) {
            // rs += dproj(act1)
            conv_w(dproj_w, w1, (long long)cH1 * cH0, stream);
            launch_gemm(act1, w1, dproj_b, rs, rs, nullptr, M, cH1, cH0,
                        0, 0, 0, 0, 1, 1.0f, 0, nullptr, 0, 0, OUT_F32, cS, stream);
        }
        const size_t wo = (size_t)i * cH1 * cH1;
        // Q (f32), K (bf16), V (bf16 transposed)
        conv_w(wq_w + wo, w0, (long long)cH1 * cH1, stream);
        launch_gemm(rs, w0, wq_b + (size_t)i * cH1, nullptr, q, nullptr,
                    M, cH1, cH1, 0, 0, 0, 0, 1, 1.0f, 0,
                    nullptr, 0, 0, OUT_F32, cS, stream);
        conv_w(wk_w + wo, w1, (long long)cH1 * cH1, stream);
        launch_gemm(rs, w1, wk_b + (size_t)i * cH1, nullptr, nullptr, k16,
                    M, cH1, cH1, 0, 0, 0, 0, 1, 1.0f, 0,
                    nullptr, 0, 0, OUT_BF16, cS, stream);
        conv_w(wv_w + wo, w0, (long long)cH1 * cH1, stream);
        launch_gemm(rs, w0, wv_b + (size_t)i * cH1, nullptr, nullptr, v16,
                    M, cH1, cH1, 0, 0, 0, 0, 1, 1.0f, 0,
                    nullptr, 0, 0, OUT_BF16_T, cS, stream);
        // scores = (Q @ K^T) * scale, causal + key-pad mask, batched over B
        launch_gemm(q, k16, nullptr, nullptr, sc, nullptr, cS, cS, cH1,
                    (long long)cS * cH1, (long long)cS * cH1,
                    (long long)cS * cS, 0,
                    cB, att_scale, 0, attmask, (long long)cS, 1,
                    OUT_F32, cS, stream);
        // softmax rows (in place, fp32)
        softmax_rows<<<dim3(cB * cS), dim3(256), 0, stream>>>(sc, cS);
        // rs += P @ V   (B = v16 is [H1,S] bf16, K-contiguous)
        launch_gemm(sc, v16, nullptr, rs, rs, nullptr, cS, cH1, cS,
                    (long long)cS * cS, (long long)cH1 * cS,
                    (long long)cS * cH1, (long long)cS * cH1,
                    cB, 1.0f, 0, nullptr, 0, 0, OUT_F32, cS, stream);
        // hid = relu(rs @ mlp1_w^T + b1)
        conv_w(mlp1_w + (size_t)i * cH4 * cH1, w1, (long long)cH4 * cH1, stream);
        launch_gemm(rs, w1, mlp1_b + (size_t)i * cH4, nullptr, hid, nullptr,
                    M, cH4, cH1, 0, 0, 0, 0, 1, 1.0f, 1,
                    nullptr, 0, 0, OUT_F32, cS, stream);
        // rs += hid @ mlp2_w^T + b2
        conv_w(mlp2_w + (size_t)i * cH1 * cH4, w0, (long long)cH1 * cH4, stream);
        launch_gemm(hid, w0, mlp2_b + (size_t)i * cH1, rs, rs, nullptr,
                    M, cH1, cH4, 0, 0, 0, 0, 1, 1.0f, 0,
                    nullptr, 0, 0, OUT_F32, cS, stream);
    }

    // out = rs @ uproj_w^T + uproj_b + z
    conv_w(uproj_w, w1, (long long)cH0 * cH1, stream);
    launch_gemm(rs, w1, uproj_b, z, out, nullptr, M, cH0, cH1,
                0, 0, 0, 0, 1, 1.0f, 0, nullptr, 0, 0, OUT_F32, cS, stream);
}
